// EvaAttention_88038239634254
// MI455X (gfx1250) — matrix operand order repaired and checked by reading, not compiled, not run on hardware
//
#include <hip/hip_runtime.h>
#include <hip/hip_bf16.h>

typedef __bf16 bf16;
typedef __attribute__((ext_vector_type(16))) __bf16 v16bf;
typedef __attribute__((ext_vector_type(8)))  __bf16 v8bf;
typedef __attribute__((ext_vector_type(8)))  float  v8f;
typedef __attribute__((ext_vector_type(4)))  unsigned int v4u;
typedef __attribute__((ext_vector_type(8)))  int v8i;
typedef __attribute__((ext_vector_type(4)))  int v4i;

#define DEV __device__ __forceinline__

#if defined(__HIP_DEVICE_COMPILE__) && __has_builtin(__builtin_amdgcn_tensor_load_to_lds)
#define HAVE_TDM 1
#else
#define HAVE_TDM 0
#endif

namespace cfg {
constexpr int  Bsz = 2, S = 8192, HID = 2048, NH = 16, HD = 128;
constexpr int  W = 256, Cc = 128, G = S / W, NC = S / Cc;
constexpr int  M = Bsz * S;                       // 16384 activation rows
constexpr float SCALE = 0.08838834764831845f;     // 128^-0.5
constexpr float MASK_MIN = -1.0e11f;
}

DEV v8f vzero() {
  v8f r;
#pragma unroll
  for (int i = 0; i < 8; ++i) r[i] = 0.f;
  return r;
}

DEV v16bf pack16(v8bf lo, v8bf hi) {
  v16bf r;
#pragma unroll
  for (int i = 0; i < 8; ++i) { r[i] = lo[i]; r[i + 8] = hi[i]; }
  return r;
}

DEV v8f wmma_bf16(v16bf a, v16bf b, v8f c) {
  // D = A(16x32 bf16) x B(32x16 bf16) + C(16x16 f32)
  return __builtin_amdgcn_wmma_f32_16x16x32_bf16(false, a, false, b, (short)0, c,
                                                 false, false);
}

#if HAVE_TDM
// TDM: DMA one 128-row x 32-col bf16 tile (row stride Kdim elems) into LDS.
// Hardware pads 4 DWORDs after every 16 DWORDs (64B row) -> LDS row stride 80B,
// matching the bank-conflict-free 40-element stride used by the WMMA readers.
DEV void tdm_tile_128x32(unsigned lds_addr, const bf16* gaddr, int kstride) {
  unsigned long long ga = (unsigned long long)(size_t)gaddr;
  v4u g0 = {1u,                                           // count=1 (valid user D#)
            lds_addr,                                     // LDS byte address
            (unsigned)ga,                                 // global_addr[31:0]
            (unsigned)((ga >> 32) & 0x01ffffffu) | (2u << 30)};  // addr[56:32] | type=2
  v8i g1;
  g1[0] = (int)((1u << 16)        // data_size = 2 bytes
              | (1u << 20)        // pad_enable
              | (3u << 22)        // pad_interval: 16 DWORDs
              | (3u << 25));      // pad_amount:   4 DWORDs
  g1[1] = (int)(((unsigned)kstride & 0xffffu) << 16);     // tensor_dim0 lo16
  g1[2] = (int)(((unsigned)kstride >> 16) & 0xffffu);     // tensor_dim0 hi16, dim1 lo16=0
  g1[3] = (int)(16u | (32u << 16));                       // tensor_dim1=1<<20, tile_dim0=32
  g1[4] = 128;                                            // tile_dim1=128, tile_dim2=0
  g1[5] = kstride;                                        // tensor_dim0_stride lo32 (elems)
  g1[6] = 0;
  g1[7] = 0;
  v4i zz4 = {0, 0, 0, 0};
  v8i zz8;
#pragma unroll
  for (int i = 0; i < 8; ++i) zz8[i] = 0;
  // 6-arg form (clang-23 / therock-10.0 headers): groups 0..3 + extra group + cpol
  __builtin_amdgcn_tensor_load_to_lds(g0, g1, zz4, zz4, zz8, 0);
}
DEV unsigned lds_off(const void* p) {   // generic LDS ptr -> addr[31:0] (aperture rule)
  return (unsigned)(size_t)p;
}
#endif

// ---------------- f32 -> bf16 conversion ----------------
__global__ void cvt_bf16(const float* __restrict__ src, bf16* __restrict__ dst, long n) {
  long i = (long)blockIdx.x * blockDim.x + threadIdx.x;
  if (i < n) dst[i] = (bf16)src[i];
}

// ---------------- tiled WMMA GEMM: C[M,N] = A[M,K] * Bw[N,K]^T ----------------
// TDM double-buffered: wave0 DMAs tile i+1 while all 8 waves run WMMA on tile i.
template <bool OUT_BF16>
__global__ __launch_bounds__(256) void gemm_nt(const bf16* __restrict__ A,
                                               const bf16* __restrict__ Bw,
                                               void* __restrict__ Cout,
                                               int Mdim, int Ndim, int Kdim) {
  __shared__ bf16 At[2][128 * 40];   // 128 rows x 32 K, stride 40 (80B, 16B aligned)
  __shared__ bf16 Bt[2][128 * 40];
  const int nb   = Ndim >> 7;
  const int bm   = blockIdx.x / nb, bn = blockIdx.x % nb;
  const int tid  = threadIdx.x, w = tid >> 5, lane = tid & 31;
  const int half = lane >> 4, lb = lane & 15;
  const int wm   = w & 3, wn = w >> 2;            // 4 M-groups x 2 N-groups of waves

  const bf16* arow = A  + (long)(bm * 128) * Kdim;
  const bf16* brow = Bw + (long)(bn * 128) * Kdim;

  v8f acc[2][4];
#pragma unroll
  for (int i = 0; i < 2; ++i)
#pragma unroll
    for (int j = 0; j < 4; ++j) acc[i][j] = vzero();

#if HAVE_TDM
  const bool issuer = (tid < 32);                 // wave 0 drives the Tensor Data Mover
  if (issuer) {
    tdm_tile_128x32(lds_off(&At[0][0]), arow, Kdim);
    tdm_tile_128x32(lds_off(&Bt[0][0]), brow, Kdim);
  }
#endif

  for (int k0 = 0; k0 < Kdim; k0 += 32) {
    const int cur = (k0 >> 5) & 1;
#if HAVE_TDM
    const bool hasNext = (k0 + 32) < Kdim;
    if (issuer) {
      if (hasNext) {
        tdm_tile_128x32(lds_off(&At[cur ^ 1][0]), arow + k0 + 32, Kdim);
        tdm_tile_128x32(lds_off(&Bt[cur ^ 1][0]), brow + k0 + 32, Kdim);
        __builtin_amdgcn_s_wait_tensorcnt(2);     // tile k0 landed; next 2 in flight
      } else {
        __builtin_amdgcn_s_wait_tensorcnt(0);
      }
    }
    __syncthreads();                              // publish LDS tile to all waves
#else
    {  // fallback: cooperative vector loads
      const int   lrow = tid >> 1, lhalf = tid & 1;
      const bf16* ga   = arow + (long)lrow * Kdim + lhalf * 16 + k0;
      const bf16* gb   = brow + (long)lrow * Kdim + lhalf * 16 + k0;
      __syncthreads();
      *(v8bf*)&At[cur][lrow * 40 + lhalf * 16]     = *(const v8bf*)ga;
      *(v8bf*)&At[cur][lrow * 40 + lhalf * 16 + 8] = *(const v8bf*)(ga + 8);
      *(v8bf*)&Bt[cur][lrow * 40 + lhalf * 16]     = *(const v8bf*)gb;
      *(v8bf*)&Bt[cur][lrow * 40 + lhalf * 16 + 8] = *(const v8bf*)(gb + 8);
      if (k0 + 32 < Kdim) {
        __builtin_prefetch(ga + 32, 0, 0);
        __builtin_prefetch(gb + 32, 0, 0);
      }
      __syncthreads();
    }
#endif

    v16bf af[2], bfr[4];
#pragma unroll
    for (int i = 0; i < 2; ++i) {   // A frag: row = lane&15; K = half*8+{0..7}, 16+half*8+{0..7}
      const bf16* p = &At[cur][(wm * 32 + i * 16 + lb) * 40 + half * 8];
      af[i] = pack16(*(const v8bf*)p, *(const v8bf*)(p + 16));
    }
#pragma unroll
    for (int j = 0; j < 4; ++j) {   // B frag: N = lane&15; K = half*8+{0..7}, 16+half*8+{0..7}
      const bf16* p = &Bt[cur][(wn * 64 + j * 16 + lb) * 40 + half * 8];
      bfr[j] = pack16(*(const v8bf*)p, *(const v8bf*)(p + 16));
    }
#pragma unroll
    for (int i = 0; i < 2; ++i)
#pragma unroll
      for (int j = 0; j < 4; ++j) acc[i][j] = wmma_bf16(af[i], bfr[j], acc[i][j]);

    __syncthreads();   // all waves done reading buf `cur` before TDM refills it
  }

#pragma unroll
  for (int i = 0; i < 2; ++i)
#pragma unroll
    for (int j = 0; j < 4; ++j)
#pragma unroll
      for (int r = 0; r < 8; ++r) {
        int   mrow = bm * 128 + wm * 32 + i * 16 + r + 8 * half;
        int   ncol = bn * 128 + wn * 64 + j * 16 + lb;
        float val  = acc[i][j][r];
        if constexpr (OUT_BF16)
          ((bf16*)Cout)[(long)mrow * Ndim + ncol] = (bf16)val;
        else
          ((float*)Cout)[(long)mrow * Ndim + ncol] = val;
      }
}

// ---------------- RoPE on q and k (in place) ----------------
__global__ void rope_qk(bf16* __restrict__ q, bf16* __restrict__ k,
                        const float* __restrict__ cosp, const float* __restrict__ sinp) {
  using namespace cfg;
  long i = (long)blockIdx.x * blockDim.x + threadIdx.x;
  if (i >= (long)M * NH * 64) return;
  int  d0   = i & 63;
  int  h    = (i >> 6) & (NH - 1);
  long bs   = i >> 10;                  // b*S + s
  long s    = bs & (S - 1);
  long base = (bs * NH + h) * HD;
  float c0 = cosp[s * HD + d0],      s0 = sinp[s * HD + d0];
  float c1 = cosp[s * HD + d0 + 64], s1 = sinp[s * HD + d0 + 64];
  float q0 = (float)q[base + d0], q1 = (float)q[base + d0 + 64];
  q[base + d0]      = (bf16)(q0 * c0 - q1 * s0);
  q[base + d0 + 64] = (bf16)(q1 * c1 + q0 * s1);
  float k0 = (float)k[base + d0], k1 = (float)k[base + d0 + 64];
  k[base + d0]      = (bf16)(k0 * c0 - k1 * s0);
  k[base + d0 + 64] = (bf16)(k1 * c1 + k0 * s1);
}

// ---------------- per-chunk stats: rf_k_bar, rfa_per_chunk ----------------
__global__ __launch_bounds__(128) void chunk_stats(
    const bf16* __restrict__ k, const bf16* __restrict__ v,
    const float* __restrict__ muk, const float* __restrict__ phi,
    bf16* __restrict__ kbar, bf16* __restrict__ rfa) {
  using namespace cfg;
  __shared__ float lmu[Cc], lph[Cc];
  int  idx = blockIdx.x;
  int  c   = idx & (NC - 1);
  int  h   = (idx >> 6) & (NH - 1);
  int  b   = idx >> 10;
  int  i   = threadIdx.x;
  long s   = (long)c * Cc + i;
  const bf16* kr = k + (((long)b * S + s) * NH + h) * HD;
  float mu = 0.f, ph = 0.f, nr = 0.f;
#pragma unroll 4
  for (int d = 0; d < HD; ++d) {
    float kv = (float)kr[d];
    mu += muk[h * HD + d] * kv;
    ph += phi[h * HD + d] * kv;
    nr += kv * kv;
  }
  lmu[i] = mu;
  lph[i] = SCALE * (ph - 0.5f * nr);
  __syncthreads();
  float mx1 = -1e30f, mx2 = -1e30f;
  for (int j = 0; j < Cc; ++j) { mx1 = fmaxf(mx1, lmu[j]); mx2 = fmaxf(mx2, lph[j]); }
  float sm1 = 0.f, sm2 = 0.f;
  for (int j = 0; j < Cc; ++j) { sm1 += __expf(lmu[j] - mx1); sm2 += __expf(lph[j] - mx2); }
  float inv1 = 1.f / sm1, inv2 = 1.f / sm2;

  int d = i;                                   // HD == Cc == 128
  const bf16* kcol = k + (((long)b * S + (long)c * Cc) * NH + h) * HD + d;
  const bf16* vcol = v + (((long)b * S + (long)c * Cc) * NH + h) * HD + d;
  float kbv = 0.f, rfv = 0.f;
  for (int j = 0; j < Cc; ++j) {
    float wmu = __expf(lmu[j] - mx1) * inv1;
    float wph = __expf(lph[j] - mx2) * inv2;
    kbv += wmu * (float)kcol[(long)j * NH * HD];
    rfv += wph * (float)vcol[(long)j * NH * HD];
  }
  long ob = (((long)b * NH + h) * NC + c) * HD + d;
  kbar[ob] = (bf16)kbv;
  rfa[ob]  = (bf16)rfv;
}

// ---------------- fused EVA attention (flash-style, WMMA) ----------------
__global__ __launch_bounds__(256) void eva_attn(
    const bf16* __restrict__ q, const bf16* __restrict__ k, const bf16* __restrict__ v,
    const bf16* __restrict__ kbar, const bf16* __restrict__ rfa, bf16* __restrict__ out) {
  using namespace cfg;
  __shared__ bf16 Vt[128 * 40];     // transposed V/rfa tile: [d][key], stride 40
  __shared__ bf16 Pb[8][16 * 32];   // per-wave prob staging: [row16][key32]

  int idx  = blockIdx.x;
  int rb   = idx & 1;
  int g    = (idx >> 1) & (G - 1);
  int h    = (idx >> 6) & (NH - 1);
  int b    = idx >> 10;
  int tid  = threadIdx.x, w = tid >> 5, lane = tid & 31;
  int half = lane >> 4, lb = lane & 15;
  int row0 = rb * 128 + w * 16;

  // Q A-fragments for full depth HD=128 (4 x K32), kept in VGPRs
  v16bf aQ[4];
  {
    long        srow = (long)g * W + row0 + lb;
    const bf16* qp   = q + (((long)b * S + srow) * NH + h) * HD;
#pragma unroll
    for (int f = 0; f < 4; ++f)
      aQ[f] = pack16(*(const v8bf*)(qp + f * 32 + half * 8),
                     *(const v8bf*)(qp + f * 32 + 16 + half * 8));
  }

  v8f   acc[8];
  float m[8], l[8];
#pragma unroll
  for (int r = 0; r < 8; ++r) { acc[r] = vzero(); m[r] = -1e30f; l[r] = 0.f; }

  for (int t = 0; t < 10; ++t) {            // 8 local key tiles + 2 chunk tiles (32 cols each)
    const bool isChunk = t >= 8;
    const int  kb      = isChunk ? (t - 8) * 32 : t * 32;

    __syncthreads();
    {  // cooperative transposed stage of V (or rfa) tile into LDS
      int         key = tid & 31, dg = tid >> 5;
      const bf16* vp  = isChunk
          ? rfa + ((((long)b * NH + h) * NC + (kb + key)) * HD) + dg * 16
          : v + (((long)b * S + (long)g * W + kb + key) * NH + h) * HD + dg * 16;
      v8bf x0 = *(const v8bf*)vp;
      v8bf x1 = *(const v8bf*)(vp + 8);
#pragma unroll
      for (int jj = 0; jj < 8; ++jj) {
        Vt[(dg * 16 + jj) * 40 + key]     = x0[jj];
        Vt[(dg * 16 + 8 + jj) * 40 + key] = x1[jj];
      }
    }
    __syncthreads();

    // wave-uniform tile skipping (future-masked local tiles / disallowed chunks)
    const bool active = isChunk ? (kb < 2 * g) : (kb <= row0 + 15);

    if (active) {
      v8f sc[2];
#pragma unroll
      for (int ct = 0; ct < 2; ++ct) {          // scores 16 rows x 16 keys, K-depth 128
        int         n  = kb + ct * 16 + lb;
        const bf16* kp = isChunk
            ? kbar + ((((long)b * NH + h) * NC + n) * HD)
            : k + (((long)b * S + (long)g * W + n) * NH + h) * HD;
        v8f cacc = vzero();
#pragma unroll
        for (int f = 0; f < 4; ++f)
          cacc = wmma_bf16(aQ[f],
                           pack16(*(const v8bf*)(kp + f * 32 + half * 8),
                                  *(const v8bf*)(kp + f * 32 + 16 + half * 8)),
                           cacc);
        sc[ct] = cacc;
      }
#pragma unroll
      for (int ct = 0; ct < 2; ++ct)            // scale + causal / chunk mask
#pragma unroll
        for (int r = 0; r < 8; ++r) {
          float sv     = sc[ct][r] * SCALE;
          int   col    = kb + ct * 16 + lb;
          int   rowi   = row0 + r + 8 * half;
          bool  masked = isChunk ? (col >= 2 * g) : (col > rowi);
          sc[ct][r]    = masked ? MASK_MIN : sv;
        }
#pragma unroll
      for (int r = 0; r < 8; ++r) {             // online softmax update
        float tm = fmaxf(sc[0][r], sc[1][r]);
#pragma unroll
        for (int dd = 1; dd < 16; dd <<= 1) tm = fmaxf(tm, __shfl_xor(tm, dd, 32));
        float mn = fmaxf(m[r], tm);
        float al = __expf(m[r] - mn);
        float p0 = __expf(sc[0][r] - mn);
        float p1 = __expf(sc[1][r] - mn);
        float rs = p0 + p1;
#pragma unroll
        for (int dd = 1; dd < 16; dd <<= 1) rs += __shfl_xor(rs, dd, 32);
        l[r] = l[r] * al + rs;
        m[r] = mn;
#pragma unroll
        for (int dt = 0; dt < 8; ++dt) acc[dt][r] *= al;
        sc[0][r] = p0;
        sc[1][r] = p1;
      }
      bf16* pb = &Pb[w][0];                     // C-layout -> A-layout via LDS
#pragma unroll
      for (int ct = 0; ct < 2; ++ct)
#pragma unroll
        for (int r = 0; r < 8; ++r)
          pb[(r + 8 * half) * 32 + ct * 16 + lb] = (bf16)sc[ct][r];
    }
    __syncthreads();
    if (active) {
      const bf16* pb = &Pb[w][0];
      v16bf aP = pack16(*(const v8bf*)(pb + lb * 32 + half * 8),
                        *(const v8bf*)(pb + lb * 32 + 16 + half * 8));
#pragma unroll
      for (int dt = 0; dt < 8; ++dt) {          // acc += P(16x32) x V(32x16)
        const bf16* bp = &Vt[(dt * 16 + lb) * 40 + half * 8];
        acc[dt] = wmma_bf16(aP, pack16(*(const v8bf*)bp, *(const v8bf*)(bp + 16)), acc[dt]);
      }
    }
  }

#pragma unroll
  for (int dt = 0; dt < 8; ++dt)                // finalize: out = acc / l
#pragma unroll
    for (int r = 0; r < 8; ++r) {
      long srow = (long)g * W + row0 + r + 8 * half;
      int  d    = dt * 16 + lb;
      out[(((long)b * S + srow) * NH + h) * HD + d] = (bf16)(acc[dt][r] / l[r]);
    }
}

// ---------------- launcher ----------------
extern "C" void kernel_launch(void* const* d_in, const int* in_sizes, int n_in,
                              void* d_out, int out_size, void* d_ws, size_t ws_size,
                              hipStream_t stream) {
  using namespace cfg;
  (void)in_sizes; (void)n_in; (void)out_size; (void)ws_size;
  const float* hidden = (const float*)d_in[0];
  const float* wq     = (const float*)d_in[1];
  const float* wk     = (const float*)d_in[2];
  const float* wv     = (const float*)d_in[3];
  const float* wo     = (const float*)d_in[4];
  const float* phi    = (const float*)d_in[5];
  const float* muk    = (const float*)d_in[6];
  const float* cosp   = (const float*)d_in[7];
  const float* sinp   = (const float*)d_in[8];

  char* ws    = (char*)d_ws;
  auto  alloc = [&](size_t bytes) {
    char* p = ws;
    ws += (bytes + 255) & ~size_t(255);
    return p;
  };
  const long NE  = (long)M * HID;
  const long WN  = (long)HID * HID;
  const long NCH = (long)Bsz * NH * NC * HD;
  bf16* hbf  = (bf16*)alloc(NE * 2);
  bf16* wqb  = (bf16*)alloc(WN * 2);
  bf16* wkb  = (bf16*)alloc(WN * 2);
  bf16* wvb  = (bf16*)alloc(WN * 2);
  bf16* wob  = (bf16*)alloc(WN * 2);
  bf16* qb   = (bf16*)alloc(NE * 2);
  bf16* kbuf = (bf16*)alloc(NE * 2);
  bf16* vb   = (bf16*)alloc(NE * 2);
  bf16* aob  = (bf16*)alloc(NE * 2);
  bf16* kbar = (bf16*)alloc(NCH * 2);
  bf16* rfab = (bf16*)alloc(NCH * 2);

  auto cdiv = [](long a, long b) { return (unsigned)((a + b - 1) / b); };
  cvt_bf16<<<cdiv(NE, 256), 256, 0, stream>>>(hidden, hbf, NE);
  cvt_bf16<<<cdiv(WN, 256), 256, 0, stream>>>(wq, wqb, WN);
  cvt_bf16<<<cdiv(WN, 256), 256, 0, stream>>>(wk, wkb, WN);
  cvt_bf16<<<cdiv(WN, 256), 256, 0, stream>>>(wv, wvb, WN);
  cvt_bf16<<<cdiv(WN, 256), 256, 0, stream>>>(wo, wob, WN);

  dim3 gg((M / 128) * (HID / 128));
  gemm_nt<true><<<gg, 256, 0, stream>>>(hbf, wqb, qb, M, HID, HID);
  gemm_nt<true><<<gg, 256, 0, stream>>>(hbf, wkb, kbuf, M, HID, HID);
  gemm_nt<true><<<gg, 256, 0, stream>>>(hbf, wvb, vb, M, HID, HID);

  rope_qk<<<cdiv((long)M * NH * 64, 256), 256, 0, stream>>>(qb, kbuf, cosp, sinp);
  chunk_stats<<<Bsz * NH * NC, 128, 0, stream>>>(kbuf, vb, muk, phi, kbar, rfab);
  eva_attn<<<Bsz * NH * G * 2, 256, 0, stream>>>(qb, kbuf, vb, kbar, rfab, aob);

  gemm_nt<false><<<gg, 256, 0, stream>>>(aob, wob, d_out, M, HID, HID);
}
